// Model_78898549227585
// MI455X (gfx1250) — matrix operand order repaired and checked by reading, not compiled, not run on hardware
//
#include <hip/hip_runtime.h>

typedef _Float16 v16h __attribute__((ext_vector_type(16)));
typedef _Float16 v8h  __attribute__((ext_vector_type(8)));
typedef float    v8f  __attribute__((ext_vector_type(8)));

#define T_IN    5
#define T_OUTV  6
#define N_AG    5
#define N_MODES 5
#define NPB     80      // nodes per block (16 samples)
#define NWAVES  5

// packed f16 weight offsets (in halves) inside d_ws
enum {
  OFF_EMB = 0,        // 4 tiles * 1 kc * 512
  OFF_EIH = 2048,     // 12 * 2 * 512
  OFF_EHH = 14336,
  OFF_DIH = 26624,
  OFF_DHH = 38912,
  OFF_GAT = 51200,    // 4 * 2 * 512
  OFF_OW1 = 55296,
  OFF_OW2 = 59392,    // 1 * 2 * 512
  OFF_CW1 = 60416,
  OFF_CW2 = 64512,
  PACK_HALVES = 65536
};

// ---- pack f32 weights into WMMA B-layout f16 ----
// B layout (v_wmma_*_16x16x32, K x N): lane holds column lane&15,
// element e -> k = kc*32 + 8*(lane>=16) + (e<8 ? e : 16 + (e-8))   (same k order as A)
__global__ void pack_kernel(const float* __restrict__ src, _Float16* __restrict__ dst,
                            int Kreal, int Nreal, int NT, int KC, int transposed, int srcLd) {
  int idx = blockIdx.x * blockDim.x + threadIdx.x;
  int total = NT * KC * 512;
  if (idx >= total) return;
  int e    = idx & 15;
  int lane = (idx >> 4) & 31;
  int t    = idx >> 9;          // tile = nt*KC + kc
  int kc   = t % KC;
  int nt   = t / KC;
  int k = kc * 32 + ((lane >> 4) << 3) + (e & 7) + ((e >> 3) << 4);
  int n = nt * 16 + (lane & 15);
  float v = 0.0f;
  if (k < Kreal && n < Nreal)
    v = transposed ? src[n * srcLd + k] : src[k * srcLd + n];
  dst[idx] = (_Float16)v;
}

// ---- WMMA helpers ----
__device__ __forceinline__ v8f wmma16(v16h a, v16h b, v8f c) {
  return __builtin_amdgcn_wmma_f32_16x16x32_f16(false, a, false, b, (short)0, c, false, false);
}
__device__ __forceinline__ v8f splat8(float v) {
  v8f r;
#pragma unroll
  for (int i = 0; i < 8; ++i) r[i] = v;
  return r;
}
__device__ __forceinline__ v8f sig8(v8f x) {
  v8f r;
#pragma unroll
  for (int i = 0; i < 8; ++i) r[i] = 1.0f / (1.0f + expf(-x[i]));
  return r;
}
__device__ __forceinline__ v8f tanh8(v8f x) {
  v8f r;
#pragma unroll
  for (int i = 0; i < 8; ++i) r[i] = tanhf(x[i]);
  return r;
}

// A layout (16-bit A 16x32): lane group g=lane>>4: e<8 -> k=kc*32+8g+e, e>=8 -> k=kc*32+8g+16+(e-8)
__device__ __forceinline__ v16h ldA(const _Float16* st, int row, int ko, int kc) {
  const _Float16* p = st + row * 64 + kc * 32 + ko;
  v8h lo = *(const v8h*)p;
  v8h hi = *(const v8h*)(p + 16);
  return __builtin_shufflevector(lo, hi, 0,1,2,3,4,5,6,7,8,9,10,11,12,13,14,15);
}
__device__ __forceinline__ v16h ldB(const _Float16* pk, int tile, int lane) {
  return *(const v16h*)(pk + (tile * 32 + lane) * 16);
}
// store f32 C-layout tile (cols colBase..colBase+15) to f16 row-major stage
__device__ __forceinline__ void stC16(_Float16* st, int rowBase, int hiL, int m, int colBase, v8f c) {
#pragma unroll
  for (int r = 0; r < 8; ++r)
    st[(rowBase + r + 8 * hiL) * 64 + colBase + m] = (_Float16)c[r];
}

// 2-chunk accumulate helper: acc += A(64) x B(64x16 tile nt)
__device__ __forceinline__ v8f mm64(const _Float16* pkW, int nt, int lane,
                                    v16h a0, v16h a1, v8f acc) {
  acc = wmma16(a0, ldB(pkW, nt * 2 + 0, lane), acc);
  acc = wmma16(a1, ldB(pkW, nt * 2 + 1, lane), acc);
  return acc;
}

// one GRU step, restructured per 16-column gate group j so that only one
// r/z pair is live at a time (no rr[]/zz[] register arrays)
__device__ __forceinline__ void gru_step(const _Float16* pkIh, const _Float16* pkHh,
                                         const float* bih, const float* bhh,
                                         v16h ax0, v16h ax1, v16h ah0, v16h ah1,
                                         int lane, int m, v8f h[4]) {
#pragma unroll
  for (int j = 0; j < 4; ++j) {
    // r gate (cols j*16..j*16+15): tile j
    v8f gi = splat8(bih[j * 16 + m]);
    gi = mm64(pkIh, j, lane, ax0, ax1, gi);
    v8f gh = splat8(bhh[j * 16 + m]);
    gh = mm64(pkHh, j, lane, ah0, ah1, gh);
    v8f r = sig8(gi + gh);
    // z gate: tile j+4
    gi = splat8(bih[(j + 4) * 16 + m]);
    gi = mm64(pkIh, j + 4, lane, ax0, ax1, gi);
    gh = splat8(bhh[(j + 4) * 16 + m]);
    gh = mm64(pkHh, j + 4, lane, ah0, ah1, gh);
    v8f z = sig8(gi + gh);
    // n gate: tile j+8
    gi = splat8(bih[(j + 8) * 16 + m]);
    gi = mm64(pkIh, j + 8, lane, ax0, ax1, gi);
    gh = splat8(bhh[(j + 8) * 16 + m]);
    gh = mm64(pkHh, j + 8, lane, ah0, ah1, gh);
    v8f n8 = tanh8(gi + r * gh);
    h[j] = (splat8(1.0f) - z) * n8 + z * h[j];
  }
}

__launch_bounds__(160, 4)   // 4 waves/SIMD -> <=256 VGPRs, no MSB addressing
__global__ void model_kernel(const float* __restrict__ x, const _Float16* __restrict__ pk,
                             const float* __restrict__ embed_b,
                             const float* __restrict__ att_src, const float* __restrict__ att_dst,
                             const float* __restrict__ gat_b,
                             const float* __restrict__ enc_bih, const float* __restrict__ enc_bhh,
                             const float* __restrict__ dec_bih, const float* __restrict__ dec_bhh,
                             const float* __restrict__ out_b1, const float* __restrict__ out_b2,
                             const float* __restrict__ proj_W, const float* __restrict__ proj_b,
                             const float* __restrict__ conf_b1, const float* __restrict__ conf_b2,
                             float* __restrict__ out, int trajOff) {
  __shared__ __align__(16) _Float16 st1[NPB * 64];   // A-stage 1 (x_emb / relu / dec_in)
  __shared__ __align__(16) _Float16 st2[NPB * 64];   // A-stage 2 (h / h_final)
  __shared__ __align__(16) _Float16 xlS[NPB * 64];   // GAT-transformed feats (f16)
  __shared__ __align__(16) float    hFS[NPB * 64];   // h_enc -> h_final (f32)
  __shared__ __align__(16) float    oT[NWAVES * 256];// per-wave 16x16 C-tile scratch
  __shared__ float asS[NPB], adS[NPB];
  __shared__ unsigned char mkS[NPB];

  const int tid  = threadIdx.x;
  const int lane = tid & 31;
  const int wave = tid >> 5;
  const int m    = lane & 15;
  const int hiL  = lane >> 4;
  const int ko   = hiL << 3;
  const int rowBase = wave * 16;                 // wave-private stage rows
  const int blockNodeBase = blockIdx.x * NPB;

  const int myNode = blockNodeBase + rowBase + m;  // node owned by this lane's A-row
  const int bS = myNode / N_AG, nS = myNode % N_AG;

  v8f h[4];
#pragma unroll
  for (int j = 0; j < 4; ++j) h[j] = splat8(0.0f);

  // ================= encoder: embed + GRU over T_IN steps =================
  for (int t = 0; t < T_IN; ++t) {
    asm volatile("" ::: "memory");   // keep weight-tile loads inside the loop (no LICM -> no spills)
    // A = raw x row, K padded to 32 (k<10 valid). group 0 holds k 0..7, group 1 holds k 8,9
    const float* xr = x + ((bS * T_IN + t) * N_AG + nS) * 10;
    v16h ax;
#pragma unroll
    for (int e = 0; e < 16; ++e) ax[e] = (_Float16)0.0f;
    if (hiL == 0) {
#pragma unroll
      for (int e = 0; e < 8; ++e) ax[e] = (_Float16)xr[e];
    } else {
      ax[0] = (_Float16)xr[8];
      ax[1] = (_Float16)xr[9];
    }
#pragma unroll
    for (int nt = 0; nt < 4; ++nt) {
      v8f acc = splat8(embed_b[nt * 16 + m]);
      acc = wmma16(ax, ldB(pk + OFF_EMB, nt, lane), acc);
      stC16(st1, rowBase, hiL, m, nt * 16, acc);   // x_emb (f16), wave-private rows
    }
    __builtin_amdgcn_wave_barrier();               // wave-local LDS ordering
    v16h ax0 = ldA(st1, rowBase + m, ko, 0);
    v16h ax1 = ldA(st1, rowBase + m, ko, 1);
#pragma unroll
    for (int j = 0; j < 4; ++j) stC16(st2, rowBase, hiL, m, j * 16, h[j]);
    __builtin_amdgcn_wave_barrier();
    v16h ah0 = ldA(st2, rowBase + m, ko, 0);
    v16h ah1 = ldA(st2, rowBase + m, ko, 1);
    gru_step(pk + OFF_EIH, pk + OFF_EHH, enc_bih, enc_bhh, ax0, ax1, ah0, ah1, lane, m, h);
  }

  // h_enc -> LDS f32
#pragma unroll
  for (int j = 0; j < 4; ++j)
#pragma unroll
    for (int r = 0; r < 8; ++r)
      hFS[(rowBase + r + 8 * hiL) * 64 + j * 16 + m] = h[j][r];
  __syncthreads();

  // ================= GAT: xl = x_emb_last @ gat_W (st1 still holds t=4 x_emb) =================
  {
    v16h a0 = ldA(st1, rowBase + m, ko, 0);
    v16h a1 = ldA(st1, rowBase + m, ko, 1);
#pragma unroll
    for (int nt = 0; nt < 4; ++nt) {
      v8f acc = splat8(0.0f);
      acc = mm64(pk + OFF_GAT, nt, lane, a0, a1, acc);
      stC16(xlS, rowBase, hiL, m, nt * 16, acc);
    }
  }
  __syncthreads();

  if (tid < NPB) {   // per-node: a_s, a_d, valid mask
    int node = blockNodeBase + tid;
    int b2 = node / N_AG, n2 = node % N_AG;
    const float* xr4 = x + ((b2 * T_IN + (T_IN - 1)) * N_AG + n2) * 10;
    float sm = 0.0f;
#pragma unroll
    for (int f = 0; f < 6; ++f) sm += xr4[f];
    mkS[tid] = (sm != 0.0f) ? 1 : 0;
    float s_ = 0.0f, d_ = 0.0f;
    for (int d = 0; d < 64; ++d) {
      float v = (float)xlS[tid * 64 + d];
      s_ += v * att_src[d];
      d_ += v * att_dst[d];
    }
    asS[tid] = s_; adS[tid] = d_;
  }
  __syncthreads();

  if (tid < NPB) {   // per-node: softmax-attention aggregate, h_final = h_enc + gat_out + gat_b
    int s5 = (tid / N_AG) * N_AG;
    float ad = adS[tid];
    bool  md = mkS[tid] != 0;
    float al[N_AG]; bool vl[N_AG]; float amax = -1e30f;
#pragma unroll
    for (int j = 0; j < N_AG; ++j) {
      int src = s5 + j;
      bool v = (src == tid) || ((mkS[src] != 0) && md);
      float a = asS[src] + ad;
      a = (a >= 0.0f) ? a : 0.2f * a;
      vl[j] = v; al[j] = a;
      if (v && a > amax) amax = a;
    }
    float w[N_AG], den = 0.0f;
#pragma unroll
    for (int j = 0; j < N_AG; ++j) { w[j] = vl[j] ? expf(al[j] - amax) : 0.0f; den += w[j]; }
    float inv = 1.0f / den;
    for (int d = 0; d < 64; ++d) {
      float acc = 0.0f;
#pragma unroll
      for (int j = 0; j < N_AG; ++j) acc += w[j] * (float)xlS[(s5 + j) * 64 + d];
      float hv = hFS[tid * 64 + d] + acc * inv + gat_b[d];
      hFS[tid * 64 + d] = hv;
      st2[tid * 64 + d] = (_Float16)hv;   // A-stage for conf head / decoder
    }
  }
  __syncthreads();

  // reload h_final into C-tiles for the decoder
#pragma unroll
  for (int j = 0; j < 4; ++j)
#pragma unroll
    for (int r = 0; r < 8; ++r)
      h[j][r] = hFS[(rowBase + r + 8 * hiL) * 64 + j * 16 + m];

  // ================= confidence head =================
  {
    v16h a0 = ldA(st2, rowBase + m, ko, 0);
    v16h a1 = ldA(st2, rowBase + m, ko, 1);
#pragma unroll
    for (int nt = 0; nt < 4; ++nt) {
      v8f acc = splat8(conf_b1[nt * 16 + m]);
      acc = mm64(pk + OFF_CW1, nt, lane, a0, a1, acc);
#pragma unroll
      for (int r = 0; r < 8; ++r)
        st1[(rowBase + r + 8 * hiL) * 64 + nt * 16 + m] = (_Float16)fmaxf(acc[r], 0.0f);
    }
    __builtin_amdgcn_wave_barrier();
    v16h c0 = ldA(st1, rowBase + m, ko, 0);
    v16h c1 = ldA(st1, rowBase + m, ko, 1);
    v8f acc = splat8((m < N_MODES) ? conf_b2[m] : 0.0f);
    acc = mm64(pk + OFF_CW2, 0, lane, c0, c1, acc);
#pragma unroll
    for (int r = 0; r < 8; ++r)
      oT[wave * 256 + (r + 8 * hiL) * 16 + m] = acc[r];
  }
  __syncthreads();

  if (tid < NPB) {   // per-node: softmax conf + initial decoder input
    int w_ = tid >> 4, row = tid & 15;
    const float* ct = &oT[w_ * 256 + row * 16];
    float mx = -1e30f;
#pragma unroll
    for (int j = 0; j < N_MODES; ++j) mx = fmaxf(mx, ct[j]);
    float e5[N_MODES], den = 0.0f;
#pragma unroll
    for (int j = 0; j < N_MODES; ++j) { e5[j] = expf(ct[j] - mx); den += e5[j]; }
    float inv = 1.0f / den;
    int node = blockNodeBase + tid;
    int b2 = node / N_AG, n2 = node % N_AG;
#pragma unroll
    for (int j = 0; j < N_MODES; ++j)
      out[trajOff + (b2 * N_MODES + j) * N_AG + n2] = e5[j] * inv;

    const float* xr4 = x + ((b2 * T_IN + (T_IN - 1)) * N_AG + n2) * 10;
    float px = xr4[0], py = xr4[1];
    for (int d = 0; d < 64; ++d)
      st1[tid * 64 + d] = (_Float16)(px * proj_W[d] + py * proj_W[64 + d] + proj_b[d]);
  }
  __syncthreads();

  // ================= decoder: T_OUT fused GRU + MLP steps =================
  for (int t = 0; t < T_OUTV; ++t) {
    asm volatile("" ::: "memory");   // keep weight-tile loads inside the loop
    v16h ax0 = ldA(st1, rowBase + m, ko, 0);   // dec input (st1)
    v16h ax1 = ldA(st1, rowBase + m, ko, 1);
#pragma unroll
    for (int j = 0; j < 4; ++j) stC16(st2, rowBase, hiL, m, j * 16, h[j]);
    __builtin_amdgcn_wave_barrier();
    v16h ah0 = ldA(st2, rowBase + m, ko, 0);
    v16h ah1 = ldA(st2, rowBase + m, ko, 1);
    gru_step(pk + OFF_DIH, pk + OFF_DHH, dec_bih, dec_bhh, ax0, ax1, ah0, ah1, lane, m, h);

    // o = relu(h @ out_W1 + b1) @ out_W2 + b2
#pragma unroll
    for (int j = 0; j < 4; ++j) stC16(st2, rowBase, hiL, m, j * 16, h[j]);
    __builtin_amdgcn_wave_barrier();
    v16h b0 = ldA(st2, rowBase + m, ko, 0);
    v16h b1 = ldA(st2, rowBase + m, ko, 1);
#pragma unroll
    for (int nt = 0; nt < 4; ++nt) {
      v8f acc = splat8(out_b1[nt * 16 + m]);
      acc = mm64(pk + OFF_OW1, nt, lane, b0, b1, acc);
#pragma unroll
      for (int r = 0; r < 8; ++r)
        st1[(rowBase + r + 8 * hiL) * 64 + nt * 16 + m] = (_Float16)fmaxf(acc[r], 0.0f);
    }
    __builtin_amdgcn_wave_barrier();
    v16h c0 = ldA(st1, rowBase + m, ko, 0);
    v16h c1 = ldA(st1, rowBase + m, ko, 1);
    v8f acc = splat8((m < 2 * N_MODES) ? out_b2[m] : 0.0f);
    acc = mm64(pk + OFF_OW2, 0, lane, c0, c1, acc);
#pragma unroll
    for (int r = 0; r < 8; ++r)
      oT[wave * 256 + (r + 8 * hiL) * 16 + m] = acc[r];
    __syncthreads();

    if (tid < NPB) {   // per-node: write trajectory, build next decoder input
      int w_ = tid >> 4, row = tid & 15;
      const float* orow = &oT[w_ * 256 + row * 16];
      int node = blockNodeBase + tid;
      int b2 = node / N_AG, n2 = node % N_AG;
#pragma unroll
      for (int mm = 0; mm < N_MODES; ++mm)
#pragma unroll
        for (int c = 0; c < 2; ++c)
          out[(((b2 * N_MODES + mm) * N_AG + n2) * T_OUTV + t) * 2 + c] = orow[c * N_MODES + mm];
      float px = orow[0], py = orow[N_MODES];   // mode-0 (x, y) fed back
      for (int d = 0; d < 64; ++d)
        st1[tid * 64 + d] = (_Float16)(px * proj_W[d] + py * proj_W[64 + d] + proj_b[d]);
    }
    __syncthreads();
  }
}

extern "C" void kernel_launch(void* const* d_in, const int* in_sizes, int n_in,
                              void* d_out, int out_size, void* d_ws, size_t ws_size,
                              hipStream_t stream) {
  (void)n_in; (void)out_size; (void)ws_size;
  const float* x       = (const float*)d_in[0];
  const float* embed_W = (const float*)d_in[1];
  const float* embed_b = (const float*)d_in[2];
  const float* gat_W   = (const float*)d_in[3];
  const float* att_src = (const float*)d_in[4];
  const float* att_dst = (const float*)d_in[5];
  const float* gat_b   = (const float*)d_in[6];
  const float* enc_Wih = (const float*)d_in[7];
  const float* enc_Whh = (const float*)d_in[8];
  const float* enc_bih = (const float*)d_in[9];
  const float* enc_bhh = (const float*)d_in[10];
  const float* dec_Wih = (const float*)d_in[11];
  const float* dec_Whh = (const float*)d_in[12];
  const float* dec_bih = (const float*)d_in[13];
  const float* dec_bhh = (const float*)d_in[14];
  const float* out_W1  = (const float*)d_in[15];
  const float* out_b1  = (const float*)d_in[16];
  const float* out_W2  = (const float*)d_in[17];
  const float* out_b2  = (const float*)d_in[18];
  const float* proj_W  = (const float*)d_in[19];
  const float* proj_b  = (const float*)d_in[20];
  const float* conf_W1 = (const float*)d_in[21];
  const float* conf_b1 = (const float*)d_in[22];
  const float* conf_W2 = (const float*)d_in[23];
  const float* conf_b2 = (const float*)d_in[24];
  float*    out = (float*)d_out;
  _Float16* pk  = (_Float16*)d_ws;

  const int B = in_sizes[0] / (T_IN * N_AG * 10);
  const int trajOff = B * N_MODES * N_AG * T_OUTV * 2;

  auto pack = [&](const float* src, int off, int Kr, int Nr, int NT, int KC, int tr, int ld) {
    int total = NT * KC * 512;
    pack_kernel<<<dim3((total + 255) / 256), dim3(256), 0, stream>>>(src, pk + off, Kr, Nr, NT, KC, tr, ld);
  };
  pack(embed_W, OFF_EMB, 10, 64, 4, 1, 0, 64);
  pack(enc_Wih, OFF_EIH, 64, 192, 12, 2, 1, 64);   // gi = x @ W_ih.T  -> B[k][n] = W[n][k]
  pack(enc_Whh, OFF_EHH, 64, 192, 12, 2, 1, 64);
  pack(dec_Wih, OFF_DIH, 64, 192, 12, 2, 1, 64);
  pack(dec_Whh, OFF_DHH, 64, 192, 12, 2, 1, 64);
  pack(gat_W,   OFF_GAT, 64, 64, 4, 2, 0, 64);
  pack(out_W1,  OFF_OW1, 64, 64, 4, 2, 0, 64);
  pack(out_W2,  OFF_OW2, 64, 10, 1, 2, 0, 10);
  pack(conf_W1, OFF_CW1, 64, 64, 4, 2, 0, 64);
  pack(conf_W2, OFF_CW2, 64, 5,  1, 2, 0, 5);

  const int blocks = (B * N_AG) / NPB;   // B = 32768 -> 2048 blocks of 160 threads (5 waves)
  model_kernel<<<dim3(blocks), dim3(NPB * 2), 0, stream>>>(
      x, pk, embed_b, att_src, att_dst, gat_b,
      enc_bih, enc_bhh, dec_bih, dec_bhh,
      out_b1, out_b2, proj_W, proj_b, conf_b1, conf_b2,
      out, trajOff);
}
